// EfficientCrossViewAttn_21466246546093
// MI455X (gfx1250) — matrix operand order repaired and checked by reading, not compiled, not run on hardware
//
#include <hip/hip_runtime.h>

// ---------- problem constants ----------
#define BB   4          // pair batch
#define VV   2          // views
#define CC   256        // channels
#define NHH  4          // heads
#define HDD  64         // head dim
#define HWW  16384      // H*W
#define GG   8          // groupnorm groups
#define GCH  32         // channels per group
#define GN_N (GCH * HWW)   // 524288 elements per (bv,group), contiguous

// ---------- tiling ----------
#define TP      32      // pixels per block
#define MROWS   64      // TP * VV rows per block
#define NT1     48      // GEMM1 N tiles (3C/16)
#define NT2     16      // GEMM2 N tiles (C/16)
#define KK8     8       // K tiles (C/32)
#define XN_STR  264     // bf16 elems; 528B row (16B aligned, dword-stride%64==4)
#define QKV_STR 776     // bf16 elems; 1552B row
#define D_STR   260     // f32 elems;  1040B row
#define SM_XN_BYTES   (MROWS * XN_STR * 2)      // 33792
#define SM_QKV_BYTES  (MROWS * QKV_STR * 2)     // 99328 (also holds xraw 64KB / D 66560B)
#define SM_BYTES      (SM_XN_BYTES + SM_QKV_BYTES)

typedef __attribute__((ext_vector_type(16))) __bf16        v16bf;
typedef __attribute__((ext_vector_type(8)))  float         v8f;
typedef __attribute__((ext_vector_type(4)))  unsigned int  v4u;

union Frag { v4u q[2]; v16bf v; };

__device__ __forceinline__ unsigned short f2bf(float f) {
    union { float f; unsigned u; } c; c.f = f;
    unsigned r = c.u + 0x7FFFu + ((c.u >> 16) & 1u);     // round-to-nearest-even
    return (unsigned short)(r >> 16);
}
__device__ __forceinline__ float bfl(unsigned u) {
    union { unsigned u; float f; } c; c.u = u << 16; return c.f;
}
__device__ __forceinline__ float bfh(unsigned u) {
    union { unsigned u; float f; } c; c.u = u & 0xFFFF0000u; return c.f;
}
// generic -> LDS byte offset (addrspacecast to AS3, then ptrtoint)
__device__ __forceinline__ unsigned ldsOffset(const void* p) {
    return (unsigned)(unsigned long long)(const __attribute__((address_space(3))) char*)p;
}

// ---------------------------------------------------------------------------
// Kernel 1: GroupNorm statistics (contiguous 2MB reduction per (bv,group))
// ---------------------------------------------------------------------------
__global__ __launch_bounds__(256) void gn_stats_kernel(const float* __restrict__ x,
                                                       float* __restrict__ stats) {
    __shared__ float2 red[256];
    const int bid = blockIdx.x;                 // 0..63 = bv*8 + g
    const float* p = x + (long)bid * GN_N;
    float s = 0.f, ss = 0.f;
    for (int i = threadIdx.x * 4; i < GN_N; i += 256 * 4) {
        float4 v = *(const float4*)(p + i);
        s  += v.x + v.y + v.z + v.w;
        ss += v.x * v.x + v.y * v.y + v.z * v.z + v.w * v.w;
    }
    red[threadIdx.x] = make_float2(s, ss);
    __syncthreads();
    for (int st = 128; st > 0; st >>= 1) {
        if (threadIdx.x < st) {
            red[threadIdx.x].x += red[threadIdx.x + st].x;
            red[threadIdx.x].y += red[threadIdx.x + st].y;
        }
        __syncthreads();
    }
    if (threadIdx.x == 0) {
        float mean = red[0].x * (1.0f / GN_N);
        float var  = red[0].y * (1.0f / GN_N) - mean * mean;
        stats[bid * 2 + 0] = mean;
        stats[bid * 2 + 1] = rsqrtf(var + 1e-5f);
    }
}

// ---------------------------------------------------------------------------
// Kernel 2: convert weights to bf16 AND swizzle into fragment-major layout:
//   frag (nt,kk) = 1KB contiguous block; within it, lane L holds the 16
//   contiguous K bf16 for o = nt*16+(L&15), kbase = kk*32+(L>>4)*16.
//   => B-fragment load becomes a fully coalesced 1KB burst.
// ---------------------------------------------------------------------------
__global__ __launch_bounds__(256) void cvt_w_kernel(const float* __restrict__ wq,
                                                    const float* __restrict__ wp,
                                                    unsigned short* __restrict__ wqt,
                                                    unsigned short* __restrict__ wpt) {
    const int idx = blockIdx.x * 256 + threadIdx.x;   // 0..16383
    if (idx < NT1 * KK8 * 32) {                       // 12288 threads for w_qkv
        const int lane = idx & 31, frag = idx >> 5;
        const int kk = frag & 7, nt = frag >> 3;
        const int o  = nt * 16 + (lane & 15);
        const int kb = kk * 32 + ((lane >> 4) << 4);
        const float* src = wq + o * CC + kb;
        unsigned short* dst = wqt + idx * 16;
        #pragma unroll
        for (int e = 0; e < 16; ++e) dst[e] = f2bf(src[e]);
    } else {                                          // 4096 threads for w_proj
        const int j = idx - NT1 * KK8 * 32;
        const int lane = j & 31, frag = j >> 5;
        const int kk = frag & 7, nt = frag >> 3;
        const int o  = nt * 16 + (lane & 15);
        const int kb = kk * 32 + ((lane >> 4) << 4);
        const float* src = wp + o * CC + kb;
        unsigned short* dst = wpt + j * 16;
        #pragma unroll
        for (int e = 0; e < 16; ++e) dst[e] = f2bf(src[e]);
    }
}

// ---------------------------------------------------------------------------
// Kernel 3: fused GroupNorm -> QKV GEMM -> 2x2 cross-view attn -> proj -> res
// ---------------------------------------------------------------------------
__global__ __launch_bounds__(256) void fused_attn_kernel(
        const float* __restrict__ x,
        const float* __restrict__ gamma, const float* __restrict__ beta,
        const float* __restrict__ b_qkv, const float* __restrict__ b_proj,
        const unsigned short* __restrict__ wqt,  // bf16, fragment-major
        const unsigned short* __restrict__ wpt,  // bf16, fragment-major
        const float* __restrict__ stats,
        float* __restrict__ out) {
    __shared__ __align__(16) unsigned char smem[SM_BYTES];
    unsigned short* xn  = (unsigned short*)smem;                    // M x XN_STR  bf16
    unsigned short* qkv = (unsigned short*)(smem + SM_XN_BYTES);    // M x QKV_STR bf16
    float*          xraw = (float*)(smem + SM_XN_BYTES);            // 512 x 32 f32 (pre-GEMM1)
    float*          Dst  = (float*)(smem + SM_XN_BYTES);            // M x D_STR f32 (post-attn)

    const int tid  = threadIdx.x;
    const int lane = tid & 31;
    const int wave = tid >> 5;
    const int b    = blockIdx.y;
    const long tilebase = (long)blockIdx.x * TP;

    // ---- Stage 1a: async DMA of raw x tile into LDS (ASYNCcnt path) ----
    // wave w owns rows [w*64, w*64+64); 16 x b128 async loads per wave
    // (4 channel-rows per instruction: lane L -> row +(L>>3), 16B at (L&7)*16)
    {
        const int r0 = wave * 64;
        #pragma unroll 4
        for (int g4 = 0; g4 < 16; ++g4) {
            const int row = r0 + g4 * 4 + (lane >> 3);     // row = u*256 + c
            const int u = row >> 8, c = row & (CC - 1);
            const unsigned voff = (unsigned)(((((long)(b * VV + u) * CC + c) * HWW
                                   + tilebase) << 2) + ((lane & 7) << 4));
            const unsigned ldst = ldsOffset(xraw) + (unsigned)(row * 128 + ((lane & 7) << 4));
            asm volatile("global_load_async_to_lds_b128 %0, %1, %2"
                         :: "v"(ldst), "v"(voff), "s"(x) : "memory");
        }
        asm volatile("s_wait_asynccnt 0x0" ::: "memory");
    }
    // ---- Stage 1b: normalize from LDS raw tile -> bf16 xn tile ----
    for (int i = wave * 64; i < wave * 64 + 64; ++i) {
        const int u = i >> 8, c = i & (CC - 1);
        const int bv = b * VV + u;
        const float mean = stats[(bv * GG + (c >> 5)) * 2 + 0];
        const float rstd = stats[(bv * GG + (c >> 5)) * 2 + 1];
        const float sc = rstd * gamma[c];
        const float sh = beta[c] - mean * sc;
        const float xv = xraw[i * 32 + lane];
        xn[(lane * 2 + u) * XN_STR + c] = f2bf(xv * sc + sh);
    }
    __syncthreads();

    // ---- Stage 2: GEMM1  qkv[m,o] = xn[m,k] * wq[o,k] + b_qkv[o] ----
    // unique N-tiles per wave (no redundant weight loads); all 4 M-tiles inner
    {
        const int koffA = (lane >> 4) << 4;              // A frag K split: 0/16
        const int arow  = lane & 15;
        const int ocol  = lane & 15;
        const int mhalf = (lane >> 4) << 3;
        for (int nt = wave; nt < NT1; nt += 8) {         // 6 N-tiles per wave
            v8f acc[4];
            #pragma unroll
            for (int mt = 0; mt < 4; ++mt) acc[mt] = (v8f){0.f,0.f,0.f,0.f,0.f,0.f,0.f,0.f};
            const unsigned short* btile = wqt + ((nt * KK8) * 32 + lane) * 16;
            #pragma unroll
            for (int kk = 0; kk < KK8; ++kk) {
                Frag Bf;
                const unsigned short* bp_ = btile + kk * 512;   // coalesced 1KB frag
                Bf.q[0] = *(const v4u*)bp_;
                Bf.q[1] = *(const v4u*)(bp_ + 8);
                #pragma unroll
                for (int mt = 0; mt < 4; ++mt) {
                    Frag Af;
                    const unsigned short* ap = xn + (mt * 16 + arow) * XN_STR + kk * 32 + koffA;
                    Af.q[0] = *(const v4u*)ap;
                    Af.q[1] = *(const v4u*)(ap + 8);
                    acc[mt] = __builtin_amdgcn_wmma_f32_16x16x32_bf16(
                                  false, Af.v, false, Bf.v, (short)0, acc[mt], false, false);
                }
            }
            const int o = nt * 16 + ocol;
            const float bias = b_qkv[o];
            #pragma unroll
            for (int mt = 0; mt < 4; ++mt)
                #pragma unroll
                for (int r = 0; r < 8; ++r)
                    qkv[(mt * 16 + mhalf + r) * QKV_STR + o] = f2bf(acc[mt][r] + bias);
        }
    }
    __syncthreads();

    // ---- Stage 3: per-pixel 2x2 cross-view attention ----
    {
        const int p  = tid >> 3;
        const int h  = (tid >> 1) & 3;
        const int qv = tid & 1;
        const unsigned short* qrow = qkv + (p * 2 + qv) * QKV_STR + h * HDD;
        const unsigned short* k0   = qkv + (p * 2 + 0) * QKV_STR + CC + h * HDD;
        const unsigned short* k1   = qkv + (p * 2 + 1) * QKV_STR + CC + h * HDD;
        float s0 = 0.f, s1 = 0.f;
        #pragma unroll 8
        for (int d = 0; d < HDD; d += 2) {
            const unsigned qu = *(const unsigned*)(qrow + d);
            const unsigned a0 = *(const unsigned*)(k0 + d);
            const unsigned a1 = *(const unsigned*)(k1 + d);
            s0 += bfl(qu) * bfl(a0) + bfh(qu) * bfh(a0);
            s1 += bfl(qu) * bfl(a1) + bfh(qu) * bfh(a1);
        }
        s0 *= 0.125f; s1 *= 0.125f;                  // hd^-0.5
        const float mx = fmaxf(s0, s1);
        const float e0 = __expf(s0 - mx), e1 = __expf(s1 - mx);
        const float inv = 1.0f / (e0 + e1);
        const float w0 = e0 * inv, w1 = e1 * inv;
        const unsigned short* v0 = qkv + (p * 2 + 0) * QKV_STR + 2 * CC + h * HDD;
        const unsigned short* v1 = qkv + (p * 2 + 1) * QKV_STR + 2 * CC + h * HDD;
        unsigned short* orow = xn + (p * 2 + qv) * XN_STR + h * HDD;
        #pragma unroll 8
        for (int d = 0; d < HDD; d += 2) {
            const unsigned u0 = *(const unsigned*)(v0 + d);
            const unsigned u1 = *(const unsigned*)(v1 + d);
            const float lo = w0 * bfl(u0) + w1 * bfl(u1);
            const float hi = w0 * bfh(u0) + w1 * bfh(u1);
            *(unsigned*)(orow + d) = (unsigned)f2bf(lo) | ((unsigned)f2bf(hi) << 16);
        }
    }
    __syncthreads();

    // ---- Stage 4: GEMM2  D[m,o] = attn[m,k] * wp[o,k] (f32 staged in LDS) --
    {
        const int koffA = (lane >> 4) << 4;
        const int arow  = lane & 15;
        const int ocol  = lane & 15;
        const int mhalf = (lane >> 4) << 3;
        for (int nt = wave; nt < NT2; nt += 8) {         // 2 N-tiles per wave
            v8f acc[4];
            #pragma unroll
            for (int mt = 0; mt < 4; ++mt) acc[mt] = (v8f){0.f,0.f,0.f,0.f,0.f,0.f,0.f,0.f};
            const unsigned short* btile = wpt + ((nt * KK8) * 32 + lane) * 16;
            #pragma unroll
            for (int kk = 0; kk < KK8; ++kk) {
                Frag Bf;
                const unsigned short* bp_ = btile + kk * 512;
                Bf.q[0] = *(const v4u*)bp_;
                Bf.q[1] = *(const v4u*)(bp_ + 8);
                #pragma unroll
                for (int mt = 0; mt < 4; ++mt) {
                    Frag Af;
                    const unsigned short* ap = xn + (mt * 16 + arow) * XN_STR + kk * 32 + koffA;
                    Af.q[0] = *(const v4u*)ap;
                    Af.q[1] = *(const v4u*)(ap + 8);
                    acc[mt] = __builtin_amdgcn_wmma_f32_16x16x32_bf16(
                                  false, Af.v, false, Bf.v, (short)0, acc[mt], false, false);
                }
            }
            const int o = nt * 16 + ocol;
            #pragma unroll
            for (int mt = 0; mt < 4; ++mt)
                #pragma unroll
                for (int r = 0; r < 8; ++r)
                    Dst[(mt * 16 + mhalf + r) * D_STR + o] = acc[mt][r];
        }
    }
    __syncthreads();

    // ---- Stage 5: epilogue — bias + residual (x re-read hits L2), coalesced
    for (int i = wave; i < VV * CC; i += 8) {
        const int u = i >> 8, c = i & (CC - 1);
        const long g = ((long)(b * VV + u) * CC + c) * HWW + tilebase + lane;
        out[g] = Dst[(lane * 2 + u) * D_STR + c] + b_proj[c] + x[g];
    }
}

// ---------------------------------------------------------------------------
extern "C" void kernel_launch(void* const* d_in, const int* in_sizes, int n_in,
                              void* d_out, int out_size, void* d_ws, size_t ws_size,
                              hipStream_t stream) {
    const float* x     = (const float*)d_in[0];
    const float* gamma = (const float*)d_in[1];
    const float* beta  = (const float*)d_in[2];
    const float* wq    = (const float*)d_in[3];
    const float* bq    = (const float*)d_in[4];
    const float* wp    = (const float*)d_in[5];
    const float* bp    = (const float*)d_in[6];
    float* out = (float*)d_out;

    // workspace layout
    float*          stats = (float*)d_ws;                                        //  512 B
    unsigned short* wqt   = (unsigned short*)((char*)d_ws + 1024);               // 384 KB
    unsigned short* wpt   = (unsigned short*)((char*)d_ws + 1024 + 3 * CC * CC * 2); // 128 KB

    gn_stats_kernel<<<BB * VV * GG, 256, 0, stream>>>(x, stats);
    cvt_w_kernel<<<64, 256, 0, stream>>>(wq, wp, wqt, wpt);
    fused_attn_kernel<<<dim3(HWW / TP, BB), 256, 0, stream>>>(
        x, gamma, beta, bq, bp, wqt, wpt, stats, out);
}
